// MSAModuleBlock_60636348284985
// MI455X (gfx1250) — matrix operand order repaired and checked by reading, not compiled, not run on hardware
//
#include <hip/hip_runtime.h>
#include <hip/hip_bf16.h>

// ---------------------------------------------------------------------------
// MSA module block for gfx1250 (MI455X). All dense contractions go through
// v_wmma_f32_16x16x32_f16 (f16 in, f32 accumulate). wave32 layout per CDNA5
// ISA 7.12.2.  Scratch requirement: ~270MB in d_ws.
//
// GEMM tiling: each wave computes a 32x(16*NT) tile (MT=2 fixed, NT in
// {1,2,4} chosen so N % (16*NT) == 0).  Per 32-deep k-step: 2*MT+2*NT b128
// loads feed MT*NT WMMAs (1.5 loads/WMMA at NT=4 vs 4.0 for naive tiling).
//
// Input index map (setup_inputs dict order, depth-first):
//  0 msa_rep[128,256,64]  1 pair_rep[256,256,128]
//  opm:   2 ln_g 3 ln_b 4 a_w 5 b_w 6 out_w 7 out_b
//  mpwa:  8 ln_g 9 ln_b 10 pln_g 11 pln_b 12 v_w 13 b_w 14 g_w 15 out_w
//  msa_t: 16 ln_g 17 ln_b 18 a_w 19 b_w 20 out_w
//  tri_out: 21..30 (ln_g ln_b ag_w a_w bg_w b_w g_w oln_g oln_b out_w)
//  tri_in:  31..40
//  attn_s:  41..48 (ln_g ln_b q_w k_w v_w b_w g_w out_w)
//  attn_e:  49..56
//  pair_t:  57 ln_g 58 ln_b 59 a_w 60 b_w 61 out_w
// Output: msa[128,256,64] then pair[256,256,128], fp32.
// ---------------------------------------------------------------------------

typedef __attribute__((ext_vector_type(16))) _Float16 v16h;
typedef __attribute__((ext_vector_type(8)))  _Float16 v8h;
typedef __attribute__((ext_vector_type(8)))  float    v8f;

#define EPI_NONE    0
#define EPI_ACCUM   1
#define EPI_MULSIG  2
#define EPI_MULSILU 3
#define EPI_SIG     4

__device__ __forceinline__ float sigf(float x) { return 1.f / (1.f + __expf(-x)); }

// ----------------------------- WMMA GEMM -----------------------------------
// C[M,N] = epilogue(alpha * A[M,K] @ BT[N,K]^T).  MT=2 x NT 16x16 tiles per
// wave, 4 waves (128 thr) per block tiling N.  Batched:
// off = (z/div)*s1 + (z%div)*s2.  Requires M%32==0 and N%(16*NT)==0.
template<int EPI, bool OH, int NT>
__global__ __launch_bounds__(128) void gemm_wmma_k(
    const _Float16* __restrict__ A, const _Float16* __restrict__ BT,
    void* __restrict__ Cv, const float* __restrict__ D, const float* __restrict__ bias,
    int M, int N, int K, int lda, int ldb, int ldc,
    long sA1, long sA2, long sB1, long sB2, long sC1, long sC2, int bdiv, float alpha)
{
    int wave = threadIdx.x >> 5;
    int lane = threadIdx.x & 31;
    int n0 = (blockIdx.x * 4 + wave) * (16 * NT);
    int m0 = blockIdx.y << 5;                   // MT=2 rows of 16
    if (n0 >= N) return;                        // wave-uniform
    int z = blockIdx.z;
    long offA = (long)(z / bdiv) * sA1 + (long)(z % bdiv) * sA2;
    long offB = (long)(z / bdiv) * sB1 + (long)(z % bdiv) * sB2;
    long offC = (long)(z / bdiv) * sC1 + (long)(z % bdiv) * sC2;
    int g = lane >> 4, r = lane & 15;
    const _Float16* Ar0 = A + offA + (long)(m0 + r) * lda;       // rows m0+r
    const _Float16* Ar1 = Ar0 + 16l * lda;                       // rows m0+16+r
    const _Float16* Brs[NT];
#pragma unroll
    for (int t = 0; t < NT; t++)
        Brs[t] = BT + offB + (long)(n0 + t * 16 + r) * ldb;      // BT row n+r
    const v8f vzero = {0.f, 0.f, 0.f, 0.f, 0.f, 0.f, 0.f, 0.f};
    v8f acc[2][NT];
#pragma unroll
    for (int t = 0; t < NT; t++) { acc[0][t] = vzero; acc[1][t] = vzero; }
    for (int k0 = 0; k0 < K; k0 += 32) {
        union { v16h v; v8h h[2]; } a0, a1, b[NT];
        // A frag: e0..7 -> k0+g*8+e ; e8..15 -> k0+16+g*8+(e-8)
        a0.h[0] = *(const v8h*)(Ar0 + k0 + g * 8);
        a0.h[1] = *(const v8h*)(Ar0 + k0 + 16 + g * 8);
        a1.h[0] = *(const v8h*)(Ar1 + k0 + g * 8);
        a1.h[1] = *(const v8h*)(Ar1 + k0 + 16 + g * 8);
        // B frag: e0..7 -> k0+g*8+e ; e8..15 -> k0+16+g*8+(e-8)
#pragma unroll
        for (int t = 0; t < NT; t++) {
            b[t].h[0] = *(const v8h*)(Brs[t] + k0 + g * 8);
            b[t].h[1] = *(const v8h*)(Brs[t] + k0 + 16 + g * 8);
        }
#pragma unroll
        for (int t = 0; t < NT; t++) {
            acc[0][t] = __builtin_amdgcn_wmma_f32_16x16x32_f16(false, a0.v, false, b[t].v,
                                                               (short)0, acc[0][t], false, false);
            acc[1][t] = __builtin_amdgcn_wmma_f32_16x16x32_f16(false, a1.v, false, b[t].v,
                                                               (short)0, acc[1][t], false, false);
        }
    }
#pragma unroll
    for (int mt = 0; mt < 2; mt++) {
#pragma unroll
        for (int t = 0; t < NT; t++) {
            int cn = n0 + t * 16 + r;
#pragma unroll
            for (int rr = 0; rr < 8; rr++) {
                int cm = m0 + mt * 16 + rr + (g << 3);
                long idx = offC + (long)cm * ldc + cn;
                float v = acc[mt][t][rr] * alpha;
                if (EPI == EPI_NONE) {
                    if (bias) v += bias[cn];
                    if (OH) ((_Float16*)Cv)[idx] = (_Float16)v; else ((float*)Cv)[idx] = v;
                } else if (EPI == EPI_ACCUM) {
                    if (bias) v += bias[cn];
                    ((float*)Cv)[idx] += v;
                } else if (EPI == EPI_MULSIG) {
                    v *= sigf(D[idx]);
                    if (OH) ((_Float16*)Cv)[idx] = (_Float16)v; else ((float*)Cv)[idx] = v;
                } else if (EPI == EPI_MULSILU) {
                    float d = D[idx]; v *= d * sigf(d);
                    if (OH) ((_Float16*)Cv)[idx] = (_Float16)v; else ((float*)Cv)[idx] = v;
                } else { // EPI_SIG
                    float sgm = sigf(v);
                    if (OH) ((_Float16*)Cv)[idx] = (_Float16)sgm; else ((float*)Cv)[idx] = sgm;
                }
            }
        }
    }
}

// ------------------------- generic gather/permute ---------------------------
// dst flat index t enumerates (d0,d1,d2,d3) row-major; src = sum(di*si).
__global__ void permute4_f2h(_Float16* __restrict__ dst, const float* __restrict__ src,
                             int D1, int D2, int D3, long s0, long s1, long s2, long s3,
                             long total) {
    long t = (long)blockIdx.x * blockDim.x + threadIdx.x;
    if (t >= total) return;
    long d3 = t % D3; long q = t / D3;
    long d2 = q % D2; q /= D2;
    long d1 = q % D1; long d0 = q / D1;
    dst[t] = (_Float16)src[d0 * s0 + d1 * s1 + d2 * s2 + d3 * s3];
}
__global__ void permute4_f2f(float* __restrict__ dst, const float* __restrict__ src,
                             int D1, int D2, int D3, long s0, long s1, long s2, long s3,
                             long total) {
    long t = (long)blockIdx.x * blockDim.x + threadIdx.x;
    if (t >= total) return;
    long d3 = t % D3; long q = t / D3;
    long d2 = q % D2; q /= D2;
    long d1 = q % D1; long d0 = q / D1;
    dst[t] = src[d0 * s0 + d1 * s1 + d2 * s2 + d3 * s3];
}

// ------------------------------ layernorm -----------------------------------
// one wave per row; C in {64,128}; f16 output (GEMM A operand)
__global__ void ln_f2h(_Float16* __restrict__ dst, const float* __restrict__ src,
                       const float* __restrict__ gam, const float* __restrict__ bet,
                       int rows, int C) {
    int wave = threadIdx.x >> 5, lane = threadIdx.x & 31;
    int row = blockIdx.x * 4 + wave;
    if (row >= rows) return;
    const float* x = src + (long)row * C;
    int per = C >> 5;
    float vals[4]; float s = 0.f;
    for (int t = 0; t < per; t++) { vals[t] = x[lane + t * 32]; s += vals[t]; }
    for (int o = 16; o > 0; o >>= 1) s += __shfl_xor(s, o, 32);
    float mean = s / (float)C;
    float ss = 0.f;
    for (int t = 0; t < per; t++) { float d = vals[t] - mean; ss += d * d; }
    for (int o = 16; o > 0; o >>= 1) ss += __shfl_xor(ss, o, 32);
    float inv = rsqrtf(ss / (float)C + 1e-5f);
    for (int t = 0; t < per; t++) {
        int c = lane + t * 32;
        dst[(long)row * C + c] = (_Float16)((vals[t] - mean) * inv * gam[c] + bet[c]);
    }
}

// ----------------------- tiny projections (N=8 / N=4) -----------------------
__global__ void proj_small(float* __restrict__ out, const _Float16* __restrict__ A,
                           const float* __restrict__ W, long rows, int K, int N) {
    long t = (long)blockIdx.x * blockDim.x + threadIdx.x;
    if (t >= rows * N) return;
    int n = (int)(t % N); long row = t / N;
    float acc = 0.f;
    for (int k = 0; k < K; k++) acc += (float)A[row * K + k] * W[k * N + n];
    out[t] = acc;
}

// ---------------- softmax over j of wtraw[i,j,h] -> wts[h][i][j] ------------
__global__ void softmax_mpwa(_Float16* __restrict__ dst, const float* __restrict__ src,
                             int n, int H) {
    int wave = threadIdx.x >> 5, lane = threadIdx.x & 31;
    int id = blockIdx.x * 4 + wave;
    if (id >= n * H) return;
    int i = id / H, h = id % H;
    float v[8]; float mx = -3.4e38f;
    for (int t = 0; t < 8; t++) {
        int j = lane + t * 32;
        v[t] = src[((long)i * n + j) * H + h];
        mx = fmaxf(mx, v[t]);
    }
    for (int o = 16; o > 0; o >>= 1) mx = fmaxf(mx, __shfl_xor(mx, o, 32));
    float s = 0.f;
    for (int t = 0; t < 8; t++) { v[t] = __expf(v[t] - mx); s += v[t]; }
    for (int o = 16; o > 0; o >>= 1) s += __shfl_xor(s, o, 32);
    float inv = 1.f / s;
    for (int t = 0; t < 8; t++)
        dst[((long)h * n + i) * (long)n + lane + t * 32] = (_Float16)(v[t] * inv);
}

// ---- attention softmax over k: raw[b][j][k] scaled + bias bp[j,k,h] --------
__global__ void softmax_attn(_Float16* __restrict__ dst, const float* __restrict__ raw,
                             const float* __restrict__ bp, long rows, float scale) {
    int wave = threadIdx.x >> 5, lane = threadIdx.x & 31;
    long rowid = (long)blockIdx.x * 4 + wave;
    if (rowid >= rows) return;
    int j = (int)(rowid & 255);
    int h = (int)((rowid >> 8) & 3);
    const float* r = raw + rowid * 256;
    float v[8]; float mx = -3.4e38f;
    for (int t = 0; t < 8; t++) {
        int kk = lane + t * 32;
        v[t] = r[kk] * scale + bp[((long)j * 256 + kk) * 4 + h];
        mx = fmaxf(mx, v[t]);
    }
    for (int o = 16; o > 0; o >>= 1) mx = fmaxf(mx, __shfl_xor(mx, o, 32));
    float s = 0.f;
    for (int t = 0; t < 8; t++) { v[t] = __expf(v[t] - mx); s += v[t]; }
    for (int o = 16; o > 0; o >>= 1) s += __shfl_xor(s, o, 32);
    float inv = 1.f / s;
    _Float16* w = dst + rowid * 256;
    for (int t = 0; t < 8; t++) w[lane + t * 32] = (_Float16)(v[t] * inv);
}

// ------------------------------ elementwise ---------------------------------
__global__ void gate_add(float* __restrict__ P, const float* __restrict__ G,
                         const float* __restrict__ T, long total) {
    long t = (long)blockIdx.x * blockDim.x + threadIdx.x;
    if (t < total) P[t] += G[t] * T[t];
}
// pair[i,j,c] += T[j,i,c]
__global__ void add_permT(float* __restrict__ P, const float* __restrict__ T, long total) {
    long t = (long)blockIdx.x * blockDim.x + threadIdx.x;
    if (t >= total) return;
    long c = t & 127; long j = (t >> 7) & 255; long i = t >> 15;
    P[t] += T[((j << 8) + i) * 128 + c];
}
// X[s,i,h*32+c] = G[s,i,h*32+c] * O[h][i][s][c]   (H_MSA=8, C_AVG=32)
__global__ void mpwa_gate(_Float16* __restrict__ X, const float* __restrict__ G,
                          const float* __restrict__ O, long total) {
    long t = (long)blockIdx.x * blockDim.x + threadIdx.x;
    if (t >= total) return;
    long c = t & 31, h = (t >> 5) & 7, i = (t >> 8) & 255, s = t >> 16;
    X[t] = (_Float16)(G[t] * O[(((h << 8) + i) * 128 + s) * 32 + c]);
}
// X[i,j,h*32+c] = G[i,j,h*32+c] * O[i][h][j][c]   (H_PAIR=4, C_TRIA=32)
__global__ void attn_gate(_Float16* __restrict__ X, const float* __restrict__ G,
                          const float* __restrict__ O, long total) {
    long t = (long)blockIdx.x * blockDim.x + threadIdx.x;
    if (t >= total) return;
    long c = t & 31, h = (t >> 5) & 3, j = (t >> 7) & 255, i = t >> 15;
    X[t] = (_Float16)(G[t] * O[(((i << 2) + h) * 256 + j) * 32 + c]);
}

// ============================================================================
extern "C" void kernel_launch(void* const* d_in, const int* in_sizes, int n_in,
                              void* d_out, int out_size, void* d_ws, size_t ws_size,
                              hipStream_t stream) {
    (void)in_sizes; (void)n_in; (void)out_size; (void)ws_size;
    const float* msa_in  = (const float*)d_in[0];
    const float* pair_in = (const float*)d_in[1];
    auto F = [&](int i) { return (const float*)d_in[i]; };

    char* base = (char*)d_ws;
    size_t off = 0;
    auto alloc = [&](size_t bytes) -> void* {
        off = (off + 255) & ~(size_t)255;
        void* p = base + off;
        off += bytes;
        return p;
    };
    auto EL = [](long total) { return dim3((unsigned)((total + 255) / 256)); };

    const long NPAIR = 8388608;  // 256*256*128
    const long NMSA  = 2097152;  // 128*256*64

    // persistent residual state
    float* pair_cur = (float*)alloc(NPAIR * 4);
    float* msa_cur  = (float*)alloc(NMSA * 4);

    // ---- pack weights: WT[N][K] = (f16) W[K][N] ----
    auto packW = [&](int idx, int K, int N) -> _Float16* {
        _Float16* w = (_Float16*)alloc((size_t)K * N * 2);
        long total = (long)N * K;
        permute4_f2h<<<EL(total), 256, 0, stream>>>(w, F(idx), N, K, 1, 0, 1, N, 0, total);
        return w;
    };
    _Float16 *opm_aW = packW(4, 64, 32), *opm_bW = packW(5, 64, 32),
             *opm_oW = packW(6, 1024, 128);
    _Float16 *mp_vW = packW(12, 64, 256), *mp_gW = packW(14, 64, 256),
             *mp_oW = packW(15, 256, 64);
    _Float16 *mt_aW = packW(18, 64, 256), *mt_bW = packW(19, 64, 256),
             *mt_oW = packW(20, 256, 64);
    _Float16 *triW[2][6];
    for (int m = 0; m < 2; m++) {
        int pb = 21 + m * 10;
        triW[m][0] = packW(pb + 2, 128, 128);  // ag
        triW[m][1] = packW(pb + 3, 128, 128);  // a
        triW[m][2] = packW(pb + 4, 128, 128);  // bg
        triW[m][3] = packW(pb + 5, 128, 128);  // b
        triW[m][4] = packW(pb + 6, 128, 128);  // g
        triW[m][5] = packW(pb + 9, 128, 128);  // out
    }
    _Float16 *attW[2][5];
    for (int m = 0; m < 2; m++) {
        int pb = 41 + m * 8;
        attW[m][0] = packW(pb + 2, 128, 128);  // q
        attW[m][1] = packW(pb + 3, 128, 128);  // k
        attW[m][2] = packW(pb + 4, 128, 128);  // v
        attW[m][3] = packW(pb + 6, 128, 128);  // g
        attW[m][4] = packW(pb + 7, 128, 128);  // out
    }
    _Float16 *pt_aW = packW(59, 128, 512), *pt_bW = packW(60, 128, 512),
             *pt_oW = packW(61, 512, 128);

    hipMemcpyAsync(pair_cur, pair_in, NPAIR * 4, hipMemcpyDeviceToDevice, stream);
    hipMemcpyAsync(msa_cur, msa_in, NMSA * 4, hipMemcpyDeviceToDevice, stream);

    size_t mark = off;  // scratch arena start (reset each phase)

    auto gemmB = [&](const _Float16* A, const _Float16* BT, void* C, const float* D,
                     const float* bias, int M, int N, int K, int lda, int ldb, int ldc,
                     long sA1, long sA2, long sB1, long sB2, long sC1, long sC2,
                     int bdiv, int batches, float alpha, int epi, bool oh) {
        int NT = (N % 256 == 0) ? 4 : ((N % 128 == 0) ? 2 : 1);
        dim3 gr((unsigned)((N + 64 * NT - 1) / (64 * NT)), (unsigned)(M / 32),
                (unsigned)batches);
        dim3 bl(128);
#define GLT(E, O, T) gemm_wmma_k<E, O, T><<<gr, bl, 0, stream>>>(A, BT, C, D, bias, \
        M, N, K, lda, ldb, ldc, sA1, sA2, sB1, sB2, sC1, sC2, bdiv, alpha)
#define GL(E, O) do { if (NT == 4) GLT(E, O, 4); else if (NT == 2) GLT(E, O, 2); \
                      else GLT(E, O, 1); } while (0)
        if (oh) {
            switch (epi) {
                case EPI_NONE: GL(EPI_NONE, true); break;
                case EPI_MULSIG: GL(EPI_MULSIG, true); break;
                case EPI_MULSILU: GL(EPI_MULSILU, true); break;
                default: GL(EPI_SIG, true); break;
            }
        } else {
            switch (epi) {
                case EPI_NONE: GL(EPI_NONE, false); break;
                case EPI_ACCUM: GL(EPI_ACCUM, false); break;
                case EPI_MULSIG: GL(EPI_MULSIG, false); break;
                case EPI_MULSILU: GL(EPI_MULSILU, false); break;
                default: GL(EPI_SIG, false); break;
            }
        }
#undef GL
#undef GLT
    };
    auto gemm1 = [&](const _Float16* A, const _Float16* BT, void* C, const float* D,
                     const float* bias, int M, int N, int K, int lda, int ldb, int ldc,
                     float alpha, int epi, bool oh) {
        gemmB(A, BT, C, D, bias, M, N, K, lda, ldb, ldc, 0, 0, 0, 0, 0, 0, 1, 1,
              alpha, epi, oh);
    };
    auto ln = [&](_Float16* dst, const float* src, int gi, int rows, int C) {
        ln_f2h<<<dim3((unsigned)((rows + 3) / 4)), 128, 0, stream>>>(dst, src, F(gi),
                                                                    F(gi + 1), rows, C);
    };

    // ======================= P1: outer product mean =========================
    {
        off = mark;
        _Float16* m_h = (_Float16*)alloc(32768l * 64 * 2);
        ln(m_h, msa_in, 2, 32768, 64);
        float* a_f = (float*)alloc(32768l * 32 * 4);
        float* b_f = (float*)alloc(32768l * 32 * 4);
        gemm1(m_h, opm_aW, a_f, nullptr, nullptr, 32768, 32, 64, 64, 64, 32, 1.f, EPI_NONE, false);
        gemm1(m_h, opm_bW, b_f, nullptr, nullptr, 32768, 32, 64, 64, 64, 32, 1.f, EPI_NONE, false);
        _Float16* aT = (_Float16*)alloc(8192l * 128 * 2);
        _Float16* bT = (_Float16*)alloc(8192l * 128 * 2);
        // aT[(i*32+c)][s] = a_f[(s*256+i)*32+c]
        permute4_f2h<<<EL(1048576), 256, 0, stream>>>(aT, a_f, 32, 128, 1, 32, 1, 8192, 0, 1048576);
        permute4_f2h<<<EL(1048576), 256, 0, stream>>>(bT, b_f, 32, 128, 1, 32, 1, 8192, 0, 1048576);
        float* O_blk = (float*)alloc(1024l * 8192 * 4);
        _Float16* P_h = (_Float16*)alloc(8192l * 1024 * 2);
        for (int ch = 0; ch < 8; ch++) {
            gemm1(aT + (long)ch * 1024 * 128, bT, O_blk, nullptr, nullptr,
                  1024, 8192, 128, 128, 128, 8192, 1.f / 128.f, EPI_NONE, false);
            // P[(il*256+j)][(c*32+d)] = O_blk[(il*32+c)][(j*32+d)]
            permute4_f2h<<<EL(8388608), 256, 0, stream>>>(P_h, O_blk, 256, 32, 32,
                                                          262144, 32, 8192, 1, 8388608);
            gemm1(P_h, opm_oW, pair_cur + (long)ch * 1048576, nullptr, F(7),
                  8192, 128, 1024, 1024, 1024, 128, 1.f, EPI_ACCUM, false);
        }
    }
    // ======================= P2: MSA pair-weighted averaging ================
    {
        off = mark;
        _Float16* m_h = (_Float16*)alloc(32768l * 64 * 2);
        ln(m_h, msa_in, 8, 32768, 64);
        _Float16* z_h = (_Float16*)alloc(65536l * 128 * 2);
        ln(z_h, pair_cur, 10, 65536, 128);
        float* v_f = (float*)alloc(32768l * 256 * 4);
        float* g_f = (float*)alloc(32768l * 256 * 4);
        gemm1(m_h, mp_vW, v_f, nullptr, nullptr, 32768, 256, 64, 64, 64, 256, 1.f, EPI_NONE, false);
        gemm1(m_h, mp_gW, g_f, nullptr, nullptr, 32768, 256, 64, 64, 64, 256, 1.f, EPI_SIG, false);
        float* wtraw = (float*)alloc(65536l * 8 * 4);
        proj_small<<<EL(65536l * 8), 256, 0, stream>>>(wtraw, z_h, F(13), 65536, 128, 8);
        _Float16* wts = (_Float16*)alloc(8l * 65536 * 2);
        softmax_mpwa<<<512, 128, 0, stream>>>(wts, wtraw, 256, 8);
        _Float16* vt = (_Float16*)alloc(8388608l * 2);
        // vt[h][s][c][j] = v_f[(s*256+j)*256 + h*32+c]
        permute4_f2h<<<EL(8388608), 256, 0, stream>>>(vt, v_f, 128, 32, 256,
                                                      32, 65536, 1, 256, 8388608);
        float* O_f = (float*)alloc(8388608l * 4);
        gemmB(wts, vt, O_f, nullptr, nullptr, 256, 4096, 256, 256, 256, 4096,
              65536, 0, 1048576, 0, 1048576, 0, 1, 8, 1.f, EPI_NONE, false);
        _Float16* X = (_Float16*)alloc(8388608l * 2);
        mpwa_gate<<<EL(8388608), 256, 0, stream>>>(X, g_f, O_f, 8388608);
        gemm1(X, mp_oW, msa_cur, nullptr, nullptr, 32768, 64, 256, 256, 256, 64,
              1.f, EPI_ACCUM, false);
    }
    // ======================= P3: MSA transition =============================
    {
        off = mark;
        _Float16* m_h = (_Float16*)alloc(32768l * 64 * 2);
        ln(m_h, msa_cur, 16, 32768, 64);
        float* t1 = (float*)alloc(32768l * 256 * 4);
        gemm1(m_h, mt_aW, t1, nullptr, nullptr, 32768, 256, 64, 64, 64, 256, 1.f, EPI_NONE, false);
        _Float16* h_h = (_Float16*)alloc(32768l * 256 * 2);
        gemm1(m_h, mt_bW, h_h, t1, nullptr, 32768, 256, 64, 64, 64, 256, 1.f, EPI_MULSILU, true);
        gemm1(h_h, mt_oW, msa_cur, nullptr, nullptr, 32768, 64, 256, 256, 256, 64,
              1.f, EPI_ACCUM, false);
    }
    // ======================= P4/P5: triangle multiplication =================
    auto tri = [&](int pb, bool outgoing, _Float16** W) {
        off = mark;
        _Float16* z_h = (_Float16*)alloc(8388608l * 2);
        ln(z_h, pair_cur, pb, 65536, 128);
        float* t1 = (float*)alloc(8388608l * 4);
        float* a_f = (float*)alloc(8388608l * 4);
        float* b_f = (float*)alloc(8388608l * 4);
        float* g_f = (float*)alloc(8388608l * 4);
        _Float16* ap = (_Float16*)alloc(8388608l * 2);
        _Float16* bp = (_Float16*)alloc(8388608l * 2);
        gemm1(z_h, W[0], t1, nullptr, nullptr, 65536, 128, 128, 128, 128, 128, 1.f, EPI_NONE, false);
        gemm1(z_h, W[1], a_f, t1, nullptr, 65536, 128, 128, 128, 128, 128, 1.f, EPI_MULSIG, false);
        gemm1(z_h, W[2], t1, nullptr, nullptr, 65536, 128, 128, 128, 128, 128, 1.f, EPI_NONE, false);
        gemm1(z_h, W[3], b_f, t1, nullptr, 65536, 128, 128, 128, 128, 128, 1.f, EPI_MULSIG, false);
        gemm1(z_h, W[4], g_f, nullptr, nullptr, 65536, 128, 128, 128, 128, 128, 1.f, EPI_SIG, false);
        if (outgoing) {  // ap[c][i][k] = a_f[(i,k,c)]
            permute4_f2h<<<EL(8388608), 256, 0, stream>>>(ap, a_f, 256, 256, 1, 1, 32768, 128, 0, 8388608);
            permute4_f2h<<<EL(8388608), 256, 0, stream>>>(bp, b_f, 256, 256, 1, 1, 32768, 128, 0, 8388608);
        } else {         // ap[c][i][k] = a_f[(k,i,c)]
            permute4_f2h<<<EL(8388608), 256, 0, stream>>>(ap, a_f, 256, 256, 1, 1, 128, 32768, 0, 8388608);
            permute4_f2h<<<EL(8388608), 256, 0, stream>>>(bp, b_f, 256, 256, 1, 1, 128, 32768, 0, 8388608);
        }
        float* x_f = a_f;  // reuse (a_f dead)
        gemmB(ap, bp, x_f, nullptr, nullptr, 256, 256, 256, 256, 256, 256,
              65536, 0, 65536, 0, 65536, 0, 1, 128, 1.f, EPI_NONE, false);
        float* xt = b_f;   // reuse: xt[(i,j,c)] = x_f[c][i][j]
        permute4_f2f<<<EL(8388608), 256, 0, stream>>>(xt, x_f, 256, 128, 1, 256, 1, 65536, 0, 8388608);
        _Float16* xl = (_Float16*)t1;  // reuse
        ln(xl, xt, pb + 7, 65536, 128);
        float* t2 = x_f;   // reuse
        gemm1(xl, W[5], t2, nullptr, nullptr, 65536, 128, 128, 128, 128, 128, 1.f, EPI_NONE, false);
        gate_add<<<EL(8388608), 256, 0, stream>>>(pair_cur, g_f, t2, 8388608);
    };
    tri(21, true, triW[0]);
    tri(31, false, triW[1]);
    // ======================= P6/P7: triangle attention ======================
    auto attn = [&](const float* src_pair, int pb, _Float16** W, float* out_dst) {
        _Float16* z_h = (_Float16*)alloc(8388608l * 2);
        ln(z_h, src_pair, pb, 65536, 128);
        _Float16* q_h = (_Float16*)alloc(8388608l * 2);
        _Float16* k_h = (_Float16*)alloc(8388608l * 2);
        gemm1(z_h, W[0], q_h, nullptr, nullptr, 65536, 128, 128, 128, 128, 128, 1.f, EPI_NONE, true);
        gemm1(z_h, W[1], k_h, nullptr, nullptr, 65536, 128, 128, 128, 128, 128, 1.f, EPI_NONE, true);
        float* v_f = (float*)alloc(8388608l * 4);
        float* g_f = (float*)alloc(8388608l * 4);
        gemm1(z_h, W[2], v_f, nullptr, nullptr, 65536, 128, 128, 128, 128, 128, 1.f, EPI_NONE, false);
        gemm1(z_h, W[3], g_f, nullptr, nullptr, 65536, 128, 128, 128, 128, 128, 1.f, EPI_SIG, false);
        float* bp_f = (float*)alloc(65536l * 4 * 4);
        proj_small<<<EL(65536l * 4), 256, 0, stream>>>(bp_f, z_h, F(pb + 5), 65536, 128, 4);
        _Float16* vt = (_Float16*)alloc(8388608l * 2);
        // vt[i][h][c][k] = v_f[(i*256+k)*128 + h*32+c]
        permute4_f2h<<<EL(8388608), 256, 0, stream>>>(vt, v_f, 4, 32, 256,
                                                      32768, 32, 1, 128, 8388608);
        float* o_f = v_f;  // reuse (v dead after pack); layout [i][h][j][c]
        float* raw = (float*)alloc(8388608l * 4);
        _Float16* at_h = (_Float16*)alloc(8388608l * 2);
        for (int ic = 0; ic < 8; ic++) {
            long i0 = (long)ic * 32;
            // logits: per (i,h): q[i,:,h,:] @ k[i,:,h,:]^T   (M=j, N=kk, K=32)
            gemmB(q_h + i0 * 32768, k_h + i0 * 32768, raw, nullptr, nullptr,
                  256, 256, 32, 128, 128, 256,
                  32768, 32, 32768, 32, 262144, 65536, 4, 128, 1.f, EPI_NONE, false);
            softmax_attn<<<8192, 128, 0, stream>>>(at_h, raw, bp_f, 32768,
                                                   0.1767766953f);
            // o: per (i,h): attn @ v   (M=j, N=c=32, K=kk=256)
            gemmB(at_h, vt + i0 * 32768, o_f + i0 * 32768, nullptr, nullptr,
                  256, 32, 256, 256, 256, 32,
                  262144, 65536, 32768, 8192, 32768, 8192, 4, 128, 1.f, EPI_NONE, false);
        }
        _Float16* X = (_Float16*)raw;  // reuse
        attn_gate<<<EL(8388608), 256, 0, stream>>>(X, g_f, o_f, 8388608);
        if (out_dst)
            gemm1(X, W[4], out_dst, nullptr, nullptr, 65536, 128, 128, 128, 128, 128,
                  1.f, EPI_NONE, false);
        else
            gemm1(X, W[4], pair_cur, nullptr, nullptr, 65536, 128, 128, 128, 128, 128,
                  1.f, EPI_ACCUM, false);
    };
    off = mark;
    attn(pair_cur, 41, attW[0], nullptr);  // starting-node attention
    off = mark;
    {
        float* pt = (float*)alloc(8388608l * 4);
        // pt[(j,i,c)] = pair[(i,j,c)]
        permute4_f2f<<<EL(8388608), 256, 0, stream>>>(pt, pair_cur, 256, 128, 1,
                                                      128, 32768, 1, 0, 8388608);
        attn(pt, 49, attW[1], pt);  // out_dst reuses pt (dead after LN inside)
        add_permT<<<EL(8388608), 256, 0, stream>>>(pair_cur, pt, 8388608);
    }
    // ======================= P8: pair transition (row-chunked) ==============
    {
        off = mark;
        _Float16* m_h = (_Float16*)alloc(8388608l * 2);
        ln(m_h, pair_cur, 57, 65536, 128);
        float* t1 = (float*)alloc(16384l * 512 * 4);
        _Float16* h_h = (_Float16*)alloc(16384l * 512 * 2);
        for (long r0 = 0; r0 < 65536; r0 += 16384) {
            gemm1(m_h + r0 * 128, pt_aW, t1, nullptr, nullptr,
                  16384, 512, 128, 128, 128, 512, 1.f, EPI_NONE, false);
            gemm1(m_h + r0 * 128, pt_bW, h_h, t1, nullptr,
                  16384, 512, 128, 128, 128, 512, 1.f, EPI_MULSILU, true);
            gemm1(h_h, pt_oW, pair_cur + r0 * 128, nullptr, nullptr,
                  16384, 128, 512, 512, 512, 128, 1.f, EPI_ACCUM, false);
        }
    }
    // ======================= output =========================================
    hipMemcpyAsync(d_out, msa_cur, NMSA * 4, hipMemcpyDeviceToDevice, stream);
    hipMemcpyAsync((float*)d_out + NMSA, pair_cur, NPAIR * 4,
                   hipMemcpyDeviceToDevice, stream);
}
